// CausalSelfAttention_2405181685814
// MI455X (gfx1250) — compile-verified, matrix operand order checked by reading, not run on hardware
//
#include <hip/hip_runtime.h>
#include <hip/hip_bf16.h>

typedef __bf16  v16bf __attribute__((ext_vector_type(16)));
typedef __bf16  bf8v  __attribute__((ext_vector_type(8)));
typedef float   v8f   __attribute__((ext_vector_type(8)));

#define B_  4
#define T_  2048
#define C_  1024
#define H_  16
#define D_  64

// Build a 16x32 (or 32x16) bf16 WMMA fragment from two contiguous 16-byte runs.
// Per ISA 7.12.2, lane's elements are K in [8h, 8h+8) then [16+8h, 16+8h+8).
union frag_u { v16bf v; bf8v h[2]; };

__device__ __forceinline__ v16bf load_frag2(const __bf16* p0, const __bf16* p1) {
  frag_u f;
  f.h[0] = *(const bf8v*)p0;
  f.h[1] = *(const bf8v*)p1;
  return f.v;
}

__device__ __forceinline__ v8f wmma_bf16(const v16bf& a, const v16bf& b, const v8f& c) {
  return __builtin_amdgcn_wmma_f32_16x16x32_bf16(false, a, false, b, (short)0, c,
                                                 false, false);
}

__device__ __forceinline__ void async_copy_b128(const __bf16* gsrc, __bf16* lds_dst) {
  const unsigned lds_addr = (unsigned)(uintptr_t)lds_dst;
  const unsigned long long gaddr = (unsigned long long)(uintptr_t)gsrc;
  asm volatile("global_load_async_to_lds_b128 %0, %1, off"
               :: "v"(lds_addr), "v"(gaddr) : "memory");
}

__device__ __forceinline__ void wait_async0() {
  asm volatile("s_wait_asynccnt 0" ::: "memory");
}

// ---------------------------------------------------------------------------
// Kernel 0: one-time weight transpose+convert: w[K][N] f32 -> wt[N][K] bf16.
// Weights are reused by 64..512 GEMM blocks, so pay the transpose once and
// make GEMM B-staging a raw async b128 copy.
// ---------------------------------------------------------------------------
__global__ __launch_bounds__(256) void transpose_cvt_kernel(
    const float* __restrict__ w, __bf16* __restrict__ wt, int N, int K) {
  __shared__ __bf16 tile[32][33];  // +1 pad against bank conflicts
  const int tid   = threadIdx.x;
  const int nbase = blockIdx.x * 32;
  const int kbase = blockIdx.y * 32;
  {
    const int r = tid >> 3;           // k row in tile
    const int c = (tid & 7) * 4;      // n col
    const float4 f = *(const float4*)(w + (size_t)(kbase + r) * N + nbase + c);
    tile[r][c + 0] = (__bf16)f.x;
    tile[r][c + 1] = (__bf16)f.y;
    tile[r][c + 2] = (__bf16)f.z;
    tile[r][c + 3] = (__bf16)f.w;
  }
  __syncthreads();
  {
    const int n = tid >> 3;
    const int c = (tid & 7) * 4;      // k col
    __bf16* o = wt + (size_t)(nbase + n) * K + kbase + c;
    o[0] = tile[c + 0][n];
    o[1] = tile[c + 1][n];
    o[2] = tile[c + 2][n];
    o[3] = tile[c + 3][n];
  }
}

// ---------------------------------------------------------------------------
// Kernel 1: qkv = x @ w_attn + b_attn.
// M = B*T = 8192, K = 1024, N = 3C = 3072. Block: 256 threads (8 waves).
// Block tile 128(M) x 64(N); wave w owns rows [w*16, w*16+16) x all 64 N.
// Software-pipelined: double-buffered LDS; B staged via async b128 copy
// (ASYNCcnt), next A tile loaded into registers while WMMAs run.
// Q,K written [B,H,T,D]; V written transposed [B,H,D,T] for the PV stage.
// ---------------------------------------------------------------------------
__global__ __launch_bounds__(256) void qkv_gemm_kernel(
    const float* __restrict__ x, const __bf16* __restrict__ wt,
    const float* __restrict__ bias, __bf16* __restrict__ qo,
    __bf16* __restrict__ ko, __bf16* __restrict__ vt) {
  __shared__ __bf16 sA[2][128 * 32];   // [m][k]  2 x 8 KB
  __shared__ __bf16 sBt[2][64 * 32];   // [n][k]  2 x 4 KB
  const int tid  = threadIdx.x;
  const int wave = tid >> 5;
  const int lane = tid & 31;
  const int mrow = lane & 15;
  const int half = lane >> 4;
  const int m_base = blockIdx.y * 128;
  const int n_blk  = blockIdx.x * 64;
  const int wm     = m_base + wave * 16;

  const int bn = tid >> 2;             // B-staging row (0..63)
  const int bc = (tid & 3) * 8;        // B-staging k-chunk

  float4 areg[4];
  // ---- prologue: kick off tile 0 ----
  async_copy_b128(wt + (size_t)(n_blk + bn) * C_ + bc, &sBt[0][bn * 32 + bc]);
#pragma unroll
  for (int i = 0; i < 4; ++i) {
    const int f = tid + i * 256;
    const int r = f >> 3;
    const int c = (f & 7) * 4;
    areg[i] = *(const float4*)(x + (size_t)(m_base + r) * C_ + c);
  }

  v8f accs[4] = {{}, {}, {}, {}};
  for (int kt = 0; kt < 32; ++kt) {
    const int cur = kt & 1;
    // commit staged A registers (f32 -> bf16) into current buffer
#pragma unroll
    for (int i = 0; i < 4; ++i) {
      const int f = tid + i * 256;
      const int r = f >> 3;
      const int c = (f & 7) * 4;
      sA[cur][r * 32 + c + 0] = (__bf16)areg[i].x;
      sA[cur][r * 32 + c + 1] = (__bf16)areg[i].y;
      sA[cur][r * 32 + c + 2] = (__bf16)areg[i].z;
      sA[cur][r * 32 + c + 3] = (__bf16)areg[i].w;
    }
    wait_async0();       // B tile for `cur` complete (own wave)
    __syncthreads();     // everyone's stores + asyncs visible; prev reads done
    if (kt + 1 < 32) {   // issue next tile while WMMAs run
      const int k1 = (kt + 1) * 32;
      async_copy_b128(wt + (size_t)(n_blk + bn) * C_ + k1 + bc,
                      &sBt[cur ^ 1][bn * 32 + bc]);
#pragma unroll
      for (int i = 0; i < 4; ++i) {
        const int f = tid + i * 256;
        const int r = f >> 3;
        const int c = (f & 7) * 4;
        areg[i] = *(const float4*)(x + (size_t)(m_base + r) * C_ + k1 + c);
      }
    }
    const __bf16* ap = &sA[cur][(wave * 16 + mrow) * 32];
    const v16bf a = load_frag2(ap + 8 * half, ap + 16 + 8 * half);
#pragma unroll
    for (int nt = 0; nt < 4; ++nt) {
      const __bf16* bp = &sBt[cur][(nt * 16 + mrow) * 32];
      const v16bf b = load_frag2(bp + 8 * half, bp + 16 + 8 * half);
      accs[nt] = wmma_bf16(a, b, accs[nt]);
    }
  }

  // Epilogue. sec is uniform per block (derived from n_blk) -> scalar branch.
  // C/D layout: col = lane%16, row = vgpr r + 8*(lane>=16).
  const int bb  = wm >> 11;               // all 16 wave rows share one batch
  const int tt0 = (wm & (T_ - 1)) + 8 * half;
#pragma unroll
  for (int nt = 0; nt < 4; ++nt) {
    const int ncol = n_blk + nt * 16;     // uniform
    const int sec  = ncol >> 10;          // 0=q 1=k 2=v
    const int n    = ncol + mrow;
    const float bv = bias[n];
    const int c    = n & (C_ - 1);
    const int h    = c >> 6;
    const int d    = c & (D_ - 1);
    if (sec == 2) {
      // V transposed [B,H,D,T]: 8 rows -> 8 consecutive t -> one b128 store
      bf8v pack;
#pragma unroll
      for (int r = 0; r < 8; ++r) pack[r] = (__bf16)(accs[nt][r] + bv);
      *(bf8v*)(vt + ((size_t)((bb * H_ + h) * D_ + d)) * T_ + tt0) = pack;
    } else {
      __bf16* dst = (sec == 0 ? qo : ko) +
                    ((size_t)((bb * H_ + h) * T_ + tt0)) * D_ + d;
#pragma unroll
      for (int r = 0; r < 8; ++r) dst[(size_t)r * D_] = (__bf16)(accs[nt][r] + bv);
    }
  }
}

// ---------------------------------------------------------------------------
// Kernel 2: flash attention. One wave per (b, h, 16-query-row tile).
// Key blocks of 32; S = Q@K^T via 4 WMMA, online softmax, P@V via 4 WMMA.
// All global fragment loads are 2x b128 per 16x32 fragment.
// ---------------------------------------------------------------------------
__global__ __launch_bounds__(256) void flash_attn_kernel(
    const __bf16* __restrict__ q, const __bf16* __restrict__ k,
    const __bf16* __restrict__ vt, __bf16* __restrict__ y) {
  __shared__ __bf16 sP[8][16 * 32];  // per-wave P repack buffer (8 KB)
  const int tid  = threadIdx.x;
  const int wave = tid >> 5;
  const int lane = tid & 31;
  const int mrow = lane & 15;
  const int half = lane >> 4;

  const int task  = blockIdx.x * 8 + wave;    // B*H*(T/16) = 8192 tasks
  const int qt    = task & 127;               // T/16 = 128
  const int h     = (task >> 7) & (H_ - 1);
  const int b     = task >> 11;
  const int qbase = qt * 16;
  const size_t headoff = ((size_t)(b * H_ + h)) * T_ * D_;
  const __bf16* vhead = vt + headoff;         // [D][T] for this head

  // Q fragments (16x64 = two K=32 chunks), A-layout, contiguous per lane
  const __bf16* qp = q + headoff + (size_t)(qbase + mrow) * D_;
  const v16bf qa0 = load_frag2(qp + 8 * half,      qp + 16 + 8 * half);
  const v16bf qa1 = load_frag2(qp + 32 + 8 * half, qp + 48 + 8 * half);

  v8f o[4] = {{}, {}, {}, {}};
  float mrun[8], lrun[8];
#pragma unroll
  for (int r = 0; r < 8; ++r) { mrun[r] = -3.0e38f; lrun[r] = 0.0f; }

  const int nblocks = (qbase + 15) / 32 + 1;
  for (int jb = 0; jb < nblocks; ++jb) {
    const int kb = jb * 32;
    if (jb + 1 < nblocks) {  // global_prefetch_b8 of the next K and V blocks
      __builtin_prefetch(k + headoff + (size_t)(kb + 32 + mrow) * D_ + 8 * half, 0, 1);
      __builtin_prefetch(vhead + (size_t)(mrow) * T_ + kb + 32 + 8 * half, 0, 1);
    }
    // ---- S = Q @ K^T : B-frag col = key, elements over d (contiguous) ----
    v8f s0 = {}, s1 = {};
#pragma unroll
    for (int nh = 0; nh < 2; ++nh) {
      const __bf16* kp = k + headoff + (size_t)(kb + nh * 16 + mrow) * D_;
      v8f sacc = {};
      {
        const v16bf bf = load_frag2(kp + 8 * half, kp + 16 + 8 * half);
        sacc = wmma_bf16(qa0, bf, sacc);
      }
      {
        const v16bf bf = load_frag2(kp + 32 + 8 * half, kp + 48 + 8 * half);
        sacc = wmma_bf16(qa1, bf, sacc);
      }
      if (nh == 0) s0 = sacc; else s1 = sacc;
    }
    // ---- scale, causal mask, online softmax ----
    float pr0[8], pr1[8], alpha[8];
#pragma unroll
    for (int r = 0; r < 8; ++r) {
      const int qi = qbase + r + 8 * half;
      float e0 = s0[r] * 0.125f;   // 1/sqrt(64)
      float e1 = s1[r] * 0.125f;
      if (kb + mrow > qi)      e0 = -3.0e38f;
      if (kb + 16 + mrow > qi) e1 = -3.0e38f;
      float rm = fmaxf(e0, e1);
#pragma unroll
      for (int off = 1; off < 16; off <<= 1) rm = fmaxf(rm, __shfl_xor(rm, off, 32));
      const float mn = fmaxf(mrun[r], rm);
      const float a  = __expf(mrun[r] - mn);
      const float p0 = __expf(e0 - mn);
      const float p1 = __expf(e1 - mn);
      float rs = p0 + p1;
#pragma unroll
      for (int off = 1; off < 16; off <<= 1) rs += __shfl_xor(rs, off, 32);
      mrun[r]  = mn;
      lrun[r]  = lrun[r] * a + rs;
      alpha[r] = a;
      pr0[r] = p0;
      pr1[r] = p1;
    }
#pragma unroll
    for (int nt = 0; nt < 4; ++nt)
#pragma unroll
      for (int r = 0; r < 8; ++r) o[nt][r] *= alpha[r];

    // ---- repack P (C-layout -> A-layout) via wave-private LDS ----
    __bf16* pbuf = &sP[wave][0];
#pragma unroll
    for (int r = 0; r < 8; ++r) {
      const int m = r + 8 * half;
      pbuf[m * 32 + mrow]      = (__bf16)pr0[r];
      pbuf[m * 32 + 16 + mrow] = (__bf16)pr1[r];
    }
    asm volatile("s_wait_dscnt 0" ::: "memory");
    const __bf16* pp = pbuf + mrow * 32;
    const v16bf pa = load_frag2(pp + 8 * half, pp + 16 + 8 * half);

    // ---- O += P @ V : V transposed [d][t] -> contiguous per lane ----
#pragma unroll
    for (int nt = 0; nt < 4; ++nt) {
      const __bf16* vp = vhead + (size_t)(nt * 16 + mrow) * T_ + kb;
      const v16bf vb = load_frag2(vp + 8 * half, vp + 16 + 8 * half);
      o[nt] = wmma_bf16(pa, vb, o[nt]);
    }
  }
  // ---- finalize: O /= l, write y in [B,T,C] bf16 ----
#pragma unroll
  for (int r = 0; r < 8; ++r) {
    const float inv = 1.0f / lrun[r];
    const int t = qbase + r + 8 * half;
    const size_t row = ((size_t)b * T_ + t) * C_ + h * D_;
#pragma unroll
    for (int nt = 0; nt < 4; ++nt)
      y[row + nt * 16 + mrow] = (__bf16)(o[nt][r] * inv);
  }
}

// ---------------------------------------------------------------------------
// Kernel 3: out = y @ w_proj + b_proj. M=8192, K=1024, N=1024, f32 output.
// Fully async-staged and software-pipelined: both A and B tiles of step
// kt+1 are in flight (ASYNCcnt) while step kt's WMMAs execute.
// ---------------------------------------------------------------------------
__global__ __launch_bounds__(256) void proj_gemm_kernel(
    const __bf16* __restrict__ y, const __bf16* __restrict__ wt,
    const float* __restrict__ bias, float* __restrict__ out) {
  __shared__ __bf16 sA[2][128 * 32];   // [m][k]
  __shared__ __bf16 sBt[2][64 * 32];   // [n][k]
  const int tid  = threadIdx.x;
  const int wave = tid >> 5;
  const int lane = tid & 31;
  const int mrow = lane & 15;
  const int half = lane >> 4;
  const int m_base = blockIdx.y * 128;
  const int n_blk  = blockIdx.x * 64;
  const int wm     = m_base + wave * 16;

  const int ar0 = (tid + 0)   >> 2, ac0 = ((tid + 0)   & 3) * 8;
  const int ar1 = (tid + 256) >> 2, ac1 = ((tid + 256) & 3) * 8;
  const int bn  = tid >> 2,         bc  = (tid & 3) * 8;

  // ---- prologue: kick off tile 0 ----
  async_copy_b128(y  + (size_t)(m_base + ar0) * C_ + ac0, &sA[0][ar0 * 32 + ac0]);
  async_copy_b128(y  + (size_t)(m_base + ar1) * C_ + ac1, &sA[0][ar1 * 32 + ac1]);
  async_copy_b128(wt + (size_t)(n_blk + bn)  * C_ + bc,  &sBt[0][bn * 32 + bc]);

  v8f accs[4] = {{}, {}, {}, {}};
  for (int kt = 0; kt < 32; ++kt) {
    const int cur = kt & 1;
    wait_async0();
    __syncthreads();
    if (kt + 1 < 32) {
      const int k1 = (kt + 1) * 32;
      async_copy_b128(y  + (size_t)(m_base + ar0) * C_ + k1 + ac0,
                      &sA[cur ^ 1][ar0 * 32 + ac0]);
      async_copy_b128(y  + (size_t)(m_base + ar1) * C_ + k1 + ac1,
                      &sA[cur ^ 1][ar1 * 32 + ac1]);
      async_copy_b128(wt + (size_t)(n_blk + bn)  * C_ + k1 + bc,
                      &sBt[cur ^ 1][bn * 32 + bc]);
    }
    const __bf16* ap = &sA[cur][(wave * 16 + mrow) * 32];
    const v16bf a = load_frag2(ap + 8 * half, ap + 16 + 8 * half);
#pragma unroll
    for (int nt = 0; nt < 4; ++nt) {
      const __bf16* bp = &sBt[cur][(nt * 16 + mrow) * 32];
      const v16bf b = load_frag2(bp + 8 * half, bp + 16 + 8 * half);
      accs[nt] = wmma_bf16(a, b, accs[nt]);
    }
  }
#pragma unroll
  for (int nt = 0; nt < 4; ++nt) {
    const int n    = n_blk + nt * 16 + mrow;
    const float bv = bias[n];
#pragma unroll
    for (int r = 0; r < 8; ++r) {
      const int m = wm + r + 8 * half;
      out[(size_t)m * C_ + n] = accs[nt][r] + bv;
    }
  }
}

// ---------------------------------------------------------------------------
extern "C" void kernel_launch(void* const* d_in, const int* in_sizes, int n_in,
                              void* d_out, int out_size, void* d_ws, size_t ws_size,
                              hipStream_t stream) {
  const float* x      = (const float*)d_in[0];
  const float* w_attn = (const float*)d_in[1];
  const float* b_attn = (const float*)d_in[2];
  const float* w_proj = (const float*)d_in[3];
  const float* b_proj = (const float*)d_in[4];
  float* out = (float*)d_out;

  const size_t BHTD = (size_t)B_ * H_ * T_ * D_;  // 8,388,608
  __bf16* qws  = (__bf16*)d_ws;
  __bf16* kws  = qws + BHTD;
  __bf16* vtws = kws + BHTD;            // V transposed [B,H,D,T]
  __bf16* yws  = vtws + BHTD;           // attention output [B,T,C]
  __bf16* wat  = yws + BHTD;            // w_attn^T bf16 [3C][C]
  __bf16* wpt  = wat + (size_t)3 * C_ * C_;  // w_proj^T bf16 [C][C]; ~76 MB total

  // 0) One-time weight convert+transpose to bf16 [n][k]
  transpose_cvt_kernel<<<dim3(96, 32), 256, 0, stream>>>(w_attn, wat, 3 * C_, C_);
  transpose_cvt_kernel<<<dim3(32, 32), 256, 0, stream>>>(w_proj, wpt, C_, C_);
  // 1) QKV projection: grid (3072/64, 8192/128), 8 waves/block
  qkv_gemm_kernel<<<dim3(48, 64), 256, 0, stream>>>(x, wat, b_attn, qws, kws, vtws);
  // 2) Flash attention: 8192 wave-tasks / 8 waves per block
  flash_attn_kernel<<<dim3(1024), 256, 0, stream>>>(qws, kws, vtws, yws);
  // 3) Output projection: grid (1024/64, 8192/128)
  proj_gemm_kernel<<<dim3(16, 64), 256, 0, stream>>>(yws, wpt, b_proj, out);
}
